// Decoder_12506944766179
// MI455X (gfx1250) — compile-verified, matrix operand order checked by reading, not run on hardware
//
#include <hip/hip_runtime.h>
#include <hip/hip_bf16.h>
#include <stdint.h>

// ---------------- problem dimensions ----------------
#define DV     10004      // vocab
#define DVP    10016      // vocab padded to 16
#define DE     512        // embedding
#define DSH    1024       // session hidden
#define DH     1024       // H
#define DB     64         // batch
#define DT     128        // seq len
#define DH2    2048       // 2*H (GRU hidden)
#define DG6    6144       // 6*H (gates)
#define PADTOK 10003

// ---------------- CDNA5 feature detection ----------------
#if defined(__has_builtin)
#  if __has_builtin(__builtin_amdgcn_tensor_load_to_lds) && \
      __has_builtin(__builtin_amdgcn_s_wait_tensorcnt)
#    define HAVE_TDM 1
#  endif
#  if __has_builtin(__builtin_amdgcn_global_load_async_to_lds_b128) && \
      __has_builtin(__builtin_amdgcn_s_wait_asynccnt)
#    define HAVE_ASYNC 1
#  endif
#endif
#ifndef HAVE_TDM
#define HAVE_TDM 0
#endif
#ifndef HAVE_ASYNC
#define HAVE_ASYNC 0
#endif
#if __has_include(<hip/amd_detail/amd_gfx1250_TDM.h>)
#define TDM_6ARG 1            // therock-10.0 headers -> 6-arg builtin
#else
#define TDM_6ARG 0            // ROCm 7.2 -> 5-arg builtin
#endif

#define ASG __attribute__((address_space(1)))
#define ASL __attribute__((address_space(3)))

typedef __attribute__((ext_vector_type(16))) __bf16 bf16x16;
typedef __attribute__((ext_vector_type(8)))  float  f32x8;
typedef __attribute__((ext_vector_type(4)))  unsigned int u32x4;
typedef __attribute__((ext_vector_type(8)))  int    i32x8;
typedef __attribute__((ext_vector_type(4)))  int    i32x4;
typedef int v4i __attribute__((vector_size(16)));   // gcc-style, matches builtin proto

union FragU { bf16x16 v; uint4 q[2]; uint16_t h[16]; };

__device__ __forceinline__ uint16_t f2bf(float f) {
  uint32_t u = __float_as_uint(f);
  u += 0x7FFFu + ((u >> 16) & 1u);          // round-to-nearest-even
  return (uint16_t)(u >> 16);
}

// A/B fragment loader for v_wmma_f32_16x16x32_bf16 from K-contiguous bf16.
// Lane l holds row r0+(l&15); element e maps to K = (e&7) + 8*((e>>3)*2 + (l>>4)),
// i.e. two contiguous 16B chunks at k0+8*(l>>4) and +16 elements further.
__device__ __forceinline__ bf16x16 load_frag(const uint16_t* __restrict__ src,
                                             int ld, int r0, int k0) {
  const int lane = threadIdx.x & 31;
  const uint16_t* p = src + (size_t)(r0 + (lane & 15)) * ld + (k0 + ((lane >> 4) << 3));
  FragU f;
  f.q[0] = *(const uint4*)(p);
  f.q[1] = *(const uint4*)(p + 16);
  return f.v;
}

// Same fragment, but from an fp32 source (LDS h-panel) with on-the-fly bf16 cvt.
__device__ __forceinline__ bf16x16 load_frag_f32(const float* __restrict__ src,
                                                 int ld, int r0, int k0) {
  const int lane = threadIdx.x & 31;
  const float* p = src + (size_t)(r0 + (lane & 15)) * ld + (k0 + ((lane >> 4) << 3));
  float4 c0 = *(const float4*)(p);
  float4 c1 = *(const float4*)(p + 4);
  float4 c2 = *(const float4*)(p + 16);
  float4 c3 = *(const float4*)(p + 20);
  FragU f;
  f.h[0]  = f2bf(c0.x); f.h[1]  = f2bf(c0.y); f.h[2]  = f2bf(c0.z); f.h[3]  = f2bf(c0.w);
  f.h[4]  = f2bf(c1.x); f.h[5]  = f2bf(c1.y); f.h[6]  = f2bf(c1.z); f.h[7]  = f2bf(c1.w);
  f.h[8]  = f2bf(c2.x); f.h[9]  = f2bf(c2.y); f.h[10] = f2bf(c2.z); f.h[11] = f2bf(c2.w);
  f.h[12] = f2bf(c3.x); f.h[13] = f2bf(c3.y); f.h[14] = f2bf(c3.z); f.h[15] = f2bf(c3.w);
  return f.v;
}

__device__ __forceinline__ f32x8 wmma_bf16(bf16x16 a, bf16x16 b, f32x8 c) {
  return __builtin_amdgcn_wmma_f32_16x16x32_bf16(false, a, false, b, (short)0, c,
                                                 false, false);
}

__device__ __forceinline__ float sigm(float x) { return 1.0f / (1.0f + __expf(-x)); }

// ---------------- conversion kernels ----------------
__global__ void k_cvt(const float* __restrict__ in, uint16_t* __restrict__ out, int n) {
  int i = blockIdx.x * blockDim.x + threadIdx.x;
  if (i < n) out[i] = f2bf(in[i]);
}

__global__ void k_cvt_wout(const float* __restrict__ in, uint16_t* __restrict__ out) {
  int i = blockIdx.x * blockDim.x + threadIdx.x;    // over DVP*DE
  if (i >= DVP * DE) return;
  int nrow = i / DE;
  out[i] = (nrow < DV) ? f2bf(in[i]) : (uint16_t)0; // zero-pad rows >= V
}

// ---------------- embedding gather ----------------
__global__ void k_embed(const int* __restrict__ x, const float* __restrict__ emb,
                        float* __restrict__ xef, uint16_t* __restrict__ xeb) {
  int row = blockIdx.x;                 // b*T + t
  int tok = x[row];
  if (tok < 0 || tok >= DV) tok = PADTOK;
  float4 v = ((const float4*)(emb + (size_t)tok * DE))[threadIdx.x];
  *(float4*)(xef + (size_t)row * DE + threadIdx.x * 4) = v;
  ushort4 u;
  u.x = f2bf(v.x); u.y = f2bf(v.y); u.z = f2bf(v.z); u.w = f2bf(v.w);
  *(ushort4*)(xeb + (size_t)row * DE + threadIdx.x * 4) = u;
}

// ---------------- h0 = cat(tanh(ses @ w1^T + b1), same) ----------------
__global__ void __launch_bounds__(128)
k_h0(const uint16_t* __restrict__ sesb, const uint16_t* __restrict__ w1b,
     const float* __restrict__ b1, float* __restrict__ h0) {
  const int wave = threadIdx.x >> 5, lane = threadIdx.x & 31;
  const int m0 = wave * 16;             // 4 waves cover all 64 batch rows
  const int n0 = blockIdx.x * 64;
  f32x8 acc[4];
  for (int s = 0; s < 4; ++s) for (int r = 0; r < 8; ++r) acc[s][r] = 0.0f;
  for (int k = 0; k < DSH; k += 32) {
    bf16x16 a = load_frag(sesb, DSH, m0, k);
#pragma unroll
    for (int s = 0; s < 4; ++s) {
      bf16x16 b = load_frag(w1b, DSH, n0 + s * 16, k);
      acc[s] = wmma_bf16(a, b, acc[s]);
    }
  }
  const int mb = m0 + ((lane >> 4) << 3);
#pragma unroll
  for (int s = 0; s < 4; ++s) {
    const int n = n0 + s * 16 + (lane & 15);
    const float bias = b1[n];
#pragma unroll
    for (int r = 0; r < 8; ++r) {
      float v = tanhf(acc[s][r] + bias);
      h0[(size_t)(mb + r) * DH2 + n]      = v;   // first half
      h0[(size_t)(mb + r) * DH2 + DH + n] = v;   // concatenated copy
    }
  }
}

// ---------------- gi = xe @ w_ih^T + b_ih  (8192 x 6144) ----------------
__global__ void __launch_bounds__(128)
k_gi(const uint16_t* __restrict__ xeb, const uint16_t* __restrict__ wihb,
     const float* __restrict__ bih, float* __restrict__ gi) {
  const int wave = threadIdx.x >> 5, lane = threadIdx.x & 31;
  const int m0 = blockIdx.x * 64 + wave * 16;
  const int n0 = blockIdx.y * 64;
  f32x8 acc[4];
  for (int s = 0; s < 4; ++s) for (int r = 0; r < 8; ++r) acc[s][r] = 0.0f;
  for (int k = 0; k < DE; k += 32) {
    bf16x16 a = load_frag(xeb, DE, m0, k);
#pragma unroll
    for (int s = 0; s < 4; ++s) {
      bf16x16 b = load_frag(wihb, DE, n0 + s * 16, k);
      acc[s] = wmma_bf16(a, b, acc[s]);
    }
  }
  const int mb = m0 + ((lane >> 4) << 3);
#pragma unroll
  for (int s = 0; s < 4; ++s) {
    const int n = n0 + s * 16 + (lane & 15);
    const float bias = bih[n];
#pragma unroll
    for (int r = 0; r < 8; ++r)
      gi[(size_t)(mb + r) * DG6 + n] = acc[s][r] + bias;
  }
}

// ---------------- one GRU step, fused gates + maxout ----------------
// WG owns hidden slice [j0, j0+32) and computes the r/z/n gate slices for it
// (M=64 batch, K=2048). h_prev is staged to LDS fp32 in 64x256 panels, moved
// by GLOBAL_LOAD_ASYNC_TO_LDS when available; the three 32x256 w_hh gate
// tiles are DMA'd into LDS by the Tensor Data Mover (2D D# descriptor).
__global__ void __launch_bounds__(128)
k_step(const float* __restrict__ hin, const uint16_t* __restrict__ whhb,
       const float* __restrict__ bhh, const float* __restrict__ gi,
       const int* __restrict__ x, int t,
       float* __restrict__ hout, uint16_t* __restrict__ decb) {
  __shared__ float shF[DB * 256];                 // 64 KB fp32 h panel
#if HAVE_TDM
  __shared__ uint16_t shW[3 * 32 * 256];          // 48 KB w_hh gate tiles
#endif
  const int tid = threadIdx.x;
  const int wave = tid >> 5, lane = tid & 31;
  const int j0 = blockIdx.x * 32;                 // hidden slice [j0, j0+32)
  const int m0 = wave * 16;

  f32x8 acc[3][2];
  for (int g = 0; g < 3; ++g)
    for (int s = 0; s < 2; ++s)
      for (int r = 0; r < 8; ++r) acc[g][s][r] = 0.0f;

  for (int kp = 0; kp < DH2; kp += 256) {
    // ---- stage 64x256 fp32 h panel into LDS ----
#if HAVE_ASYNC
    for (int i = tid * 4; i < DB * 256; i += 128 * 4) {
      int rr = i >> 8, cc = i & 255;
      __builtin_amdgcn_global_load_async_to_lds_b128(
          (ASG v4i*)(hin + (size_t)rr * DH2 + kp + cc),
          (ASL v4i*)(shF + i), 0, 0);
    }
    __builtin_amdgcn_s_wait_asynccnt(0);
#else
    for (int i = tid * 4; i < DB * 256; i += 128 * 4) {
      int rr = i >> 8, cc = i & 255;
      const float* gp = hin + (size_t)rr * DH2 + kp + cc;
      if (kp + 256 < DH2) __builtin_prefetch(gp + 256, 0, 1);
      *(float4*)(shF + i) = *(const float4*)gp;
    }
#endif
#if HAVE_TDM
    // ---- TDM: DMA the three 32x256 bf16 w_hh gate tiles into LDS ----
    if (wave == 0) {
#pragma unroll
      for (int g = 0; g < 3; ++g) {
        const uint16_t* gsrc = whhb + ((size_t)(g * DH2 + j0)) * DH2 + kp;
        uint64_t ga = (uint64_t)(uintptr_t)gsrc;
        uint32_t la = (uint32_t)(uintptr_t)(ASL uint16_t*)(shW + g * 32 * 256);
        u32x4 g0 = { 1u,                                     // count=1, load
                     la,                                     // lds_addr [63:32]
                     (uint32_t)ga,                           // global_addr lo
                     ((uint32_t)(ga >> 32) & 0x01FFFFFFu) | (2u << 30) }; // hi | type=2
        i32x8 g1 = { (int)(1u << 16),          // data_size=2B, wg_mask=0
                     (int)(2048u << 16),       // tensor_dim0 = 2048 (lo 16 into [31:16])
                     (int)(6144u << 16),       // dim0_hi=0 | tensor_dim1 = 6144
                     (int)(256u << 16),        // dim1_hi=0 | tile_dim0 = 256
                     32,                        // tile_dim1 = 32, tile_dim2 = 0
                     2048,                      // tensor_dim0_stride lo
                     0, 0 };                    // stride hi, dim1_stride (unused, 2D)
        i32x4 gz = { 0, 0, 0, 0 };
#if TDM_6ARG
        i32x8 z8 = { 0, 0, 0, 0, 0, 0, 0, 0 };
        __builtin_amdgcn_tensor_load_to_lds(g0, g1, gz, gz, z8, 0);
#else
        __builtin_amdgcn_tensor_load_to_lds(g0, g1, gz, gz, 0);
#endif
      }
      __builtin_amdgcn_s_wait_tensorcnt(0);
    }
#endif
    __syncthreads();

    for (int kk = 0; kk < 256; kk += 32) {
      bf16x16 a = load_frag_f32(shF, 256, m0, kk);   // LDS A-fragment (+cvt)
#pragma unroll
      for (int g = 0; g < 3; ++g)
#pragma unroll
        for (int s = 0; s < 2; ++s) {
#if HAVE_TDM
          bf16x16 b = load_frag(shW + g * 32 * 256, 256, s * 16, kk);
#else
          bf16x16 b = load_frag(whhb, DH2, g * DH2 + j0 + s * 16, kp + kk);
#endif
          acc[g][s] = wmma_bf16(a, b, acc[g][s]);
        }
    }
    __syncthreads();
  }

  const int mb = m0 + ((lane >> 4) << 3);
#pragma unroll
  for (int s = 0; s < 2; ++s) {
    const int n = j0 + s * 16 + (lane & 15);
    const float br = bhh[n], bz = bhh[DH2 + n], bn = bhh[2 * DH2 + n];
#pragma unroll
    for (int r = 0; r < 8; ++r) {
      const int m = mb + r;
      const size_t grow = ((size_t)m * DT + t) * DG6;
      float hr = acc[0][s][r] + br;
      float hz = acc[1][s][r] + bz;
      float hn = acc[2][s][r] + bn;
      float rr = sigm(gi[grow + n] + hr);
      float zz = sigm(gi[grow + DH2 + n] + hz);
      float nn = tanhf(gi[grow + 2 * DH2 + n] + rr * hn);
      float hp = hin[(size_t)m * DH2 + n];
      float hv = (1.0f - zz) * nn + zz * hp;
      hout[(size_t)m * DH2 + n] = hv;
      // maxout over adjacent hidden pair via wave32 lane-xor shuffle + mask
      float other = __shfl_xor(hv, 1, 32);
      if ((lane & 1) == 0) {
        float msk = (x[m * DT + t] < PADTOK) ? 1.0f : 0.0f;
        decb[((size_t)m * DT + t) * DH + (n >> 1)] = f2bf(fmaxf(hv, other) * msk);
      }
    }
  }
}

// ---------------- dec2 = maxout @ w2^T + b2 + xe  (8192 x 512, bf16 out) ----------------
__global__ void __launch_bounds__(128)
k_dec2(const uint16_t* __restrict__ decb, const uint16_t* __restrict__ w2b,
       const float* __restrict__ b2, const float* __restrict__ xef,
       uint16_t* __restrict__ dec2) {
  const int wave = threadIdx.x >> 5, lane = threadIdx.x & 31;
  const int m0 = blockIdx.x * 64 + wave * 16;
  const int n0 = blockIdx.y * 64;
  f32x8 acc[4];
  for (int s = 0; s < 4; ++s) for (int r = 0; r < 8; ++r) acc[s][r] = 0.0f;
  for (int k = 0; k < DH; k += 32) {
    bf16x16 a = load_frag(decb, DH, m0, k);
#pragma unroll
    for (int s = 0; s < 4; ++s) {
      bf16x16 b = load_frag(w2b, DH, n0 + s * 16, k);
      acc[s] = wmma_bf16(a, b, acc[s]);
    }
  }
  const int mb = m0 + ((lane >> 4) << 3);
#pragma unroll
  for (int s = 0; s < 4; ++s) {
    const int n = n0 + s * 16 + (lane & 15);
    const float bias = b2[n];
#pragma unroll
    for (int r = 0; r < 8; ++r) {
      float v = acc[s][r] + bias + xef[(size_t)(mb + r) * DE + n];
      dec2[(size_t)(mb + r) * DE + n] = f2bf(v);
    }
  }
}

// ---------------- logits = dec2 @ w_out^T  (8192 x 10004 fp32) ----------------
__global__ void __launch_bounds__(128)
k_logits(const uint16_t* __restrict__ dec2, const uint16_t* __restrict__ woutb,
         float* __restrict__ out) {
  const int wave = threadIdx.x >> 5, lane = threadIdx.x & 31;
  const int m0 = blockIdx.x * 64 + wave * 16;
  const int n0 = blockIdx.y * 64;
  f32x8 acc[4];
  for (int s = 0; s < 4; ++s) for (int r = 0; r < 8; ++r) acc[s][r] = 0.0f;
  for (int k = 0; k < DE; k += 32) {
    bf16x16 a = load_frag(dec2, DE, m0, k);
#pragma unroll
    for (int s = 0; s < 4; ++s) {
      bf16x16 b = load_frag(woutb, DE, n0 + s * 16, k);   // padded weights
      acc[s] = wmma_bf16(a, b, acc[s]);
    }
  }
  const int mb = m0 + ((lane >> 4) << 3);
#pragma unroll
  for (int s = 0; s < 4; ++s) {
    const int n = n0 + s * 16 + (lane & 15);
    if (n < DV) {
#pragma unroll
      for (int r = 0; r < 8; ++r)
        out[(size_t)(mb + r) * DV + n] = acc[s][r];
    }
  }
}

// ---------------- host-side orchestration ----------------
extern "C" void kernel_launch(void* const* d_in, const int* in_sizes, int n_in,
                              void* d_out, int out_size, void* d_ws, size_t ws_size,
                              hipStream_t stream) {
  (void)in_sizes; (void)n_in; (void)out_size; (void)ws_size;
  const float* ses  = (const float*)d_in[0];
  const int*   x    = (const int*)  d_in[1];
  // d_in[2] = x_lens (unused: full lengths)
  const float* emb  = (const float*)d_in[3];
  const float* w1   = (const float*)d_in[4];
  const float* b1   = (const float*)d_in[5];
  const float* wih  = (const float*)d_in[6];
  const float* whh  = (const float*)d_in[7];
  const float* bih  = (const float*)d_in[8];
  const float* bhh  = (const float*)d_in[9];
  const float* w2   = (const float*)d_in[10];
  const float* b2   = (const float*)d_in[11];
  const float* wout = (const float*)d_in[12];
  float* out = (float*)d_out;

  char* base = (char*)d_ws;
  size_t off = 0;
  auto alloc = [&](size_t bytes) -> void* {
    void* p = base + off;
    off = (off + bytes + 255) & ~(size_t)255;
    return p;
  };
  uint16_t* W1b   = (uint16_t*)alloc((size_t)DH  * DSH * 2);
  uint16_t* WIHb  = (uint16_t*)alloc((size_t)DG6 * DE  * 2);
  uint16_t* WHHb  = (uint16_t*)alloc((size_t)DG6 * DH2 * 2);
  uint16_t* W2b   = (uint16_t*)alloc((size_t)DE  * DH  * 2);
  uint16_t* WOUTb = (uint16_t*)alloc((size_t)DVP * DE  * 2);
  uint16_t* SESb  = (uint16_t*)alloc((size_t)DB  * DSH * 2);
  uint16_t* XEB   = (uint16_t*)alloc((size_t)DB * DT * DE * 2);
  uint16_t* DECB  = (uint16_t*)alloc((size_t)DB * DT * DH * 2);
  uint16_t* DEC2B = (uint16_t*)alloc((size_t)DB * DT * DE * 2);
  float*    XEF   = (float*)   alloc((size_t)DB * DT * DE * 4);
  float*    H0    = (float*)   alloc((size_t)DB * DH2 * 4);
  float*    H1    = (float*)   alloc((size_t)DB * DH2 * 4);
  float*    GI    = (float*)   alloc((size_t)DB * DT * DG6 * 4);

  const int thr = 256;
  k_cvt<<<(DH * DSH + thr - 1) / thr, thr, 0, stream>>>(w1, W1b, DH * DSH);
  k_cvt<<<(DG6 * DE + thr - 1) / thr, thr, 0, stream>>>(wih, WIHb, DG6 * DE);
  k_cvt<<<(DG6 * DH2 + thr - 1) / thr, thr, 0, stream>>>(whh, WHHb, DG6 * DH2);
  k_cvt<<<(DE * DH + thr - 1) / thr, thr, 0, stream>>>(w2, W2b, DE * DH);
  k_cvt<<<(DB * DSH + thr - 1) / thr, thr, 0, stream>>>(ses, SESb, DB * DSH);
  k_cvt_wout<<<(DVP * DE + thr - 1) / thr, thr, 0, stream>>>(wout, WOUTb);

  k_h0<<<DH / 64, 128, 0, stream>>>(SESb, W1b, b1, H0);
  k_embed<<<DB * DT, 128, 0, stream>>>(x, emb, XEF, XEB);
  k_gi<<<dim3((DB * DT) / 64, DG6 / 64), 128, 0, stream>>>(XEB, WIHb, bih, GI);

  for (int t = 0; t < DT; ++t) {
    const float* hi = (t & 1) ? H1 : H0;
    float*       ho = (t & 1) ? H0 : H1;
    k_step<<<DH2 / 32, 128, 0, stream>>>(hi, WHHb, bhh, GI, x, t, ho, DECB);
  }

  k_dec2<<<dim3((DB * DT) / 64, DE / 64), 128, 0, stream>>>(DECB, W2b, b2, XEF, DEC2B);
  k_logits<<<dim3((DB * DT) / 64, DVP / 64), 128, 0, stream>>>(DEC2B, WOUTb, out);
}
